// TransformerLayerController_69758858822080
// MI455X (gfx1250) — matrix operand order repaired and checked by reading, not compiled, not run on hardware
//
#include <hip/hip_runtime.h>
#include <hip/hip_bf16.h>
#include <math.h>

// ---------------- geometry ----------------
#define BH   32                 // B*H
#define SEQ  2048
#define DIM  128
#define NTOT (BH * SEQ * DIM)   // 8,388,608
#define NOUT (NTOT / 256)       // 32768 outliers
#define SINK 4
#define QMAXF 7.0f
#define SMSCALE 0.088388347648318447f  // 1/sqrt(128)

typedef __attribute__((ext_vector_type(16))) _Float16 v16h;
typedef __attribute__((ext_vector_type(8)))  float    v8f;

// ---------------- workspace layout (bytes) ----------------
//   0      : hist_k  u32[2048]
//   8192   : hist_v  u32[2048]
//   16384  : thr     u32[16]      (thr[0]=K bits, thr[1]=V bits)
//   16448  : scale_k f32[4096]
//   32832  : scale_v f32[4096]
//   65536  : KrecT   f16[BH][DIM][SEQ]   (16 MiB)  -- transposed for QK^T B-frags
//   +16MiB : Vrec    f16[BH][SEQ][DIM]   (16 MiB)  -- row-major for PV B-frags
//   +32MiB : Qpack   f16 (A-fragment lane order)   (16 MiB)

// ---------------- preprocessing ----------------
__global__ void zero_ws_kernel(unsigned int* w, int n) {
  int i = blockIdx.x * blockDim.x + threadIdx.x;
  if (i < n) w[i] = 0u;
}

__global__ void hist_kernel(const float* __restrict__ x,
                            unsigned int* __restrict__ hist, int n) {
  int i = blockIdx.x * blockDim.x + threadIdx.x;
  if (i < n) {
    unsigned int b = __float_as_uint(fabsf(x[i])) >> 21;  // monotone in |x|
    atomicAdd(&hist[b], 1u);
  }
}

// pick bin where cumulative-from-top crosses NOUT; threshold = bin lower edge
__global__ void select_thr_kernel(const unsigned int* __restrict__ hk,
                                  const unsigned int* __restrict__ hv,
                                  unsigned int* __restrict__ thr) {
  if (threadIdx.x == 0 && blockIdx.x == 0) {
    unsigned long long c = 0; unsigned int bk = 0;
    for (int b = 2047; b >= 0; --b) { c += hk[b]; if (c >= (unsigned long long)NOUT) { bk = (unsigned int)b; break; } }
    thr[0] = bk << 21;
    c = 0; unsigned int bv = 0;
    for (int b = 2047; b >= 0; --b) { c += hv[b]; if (c >= (unsigned long long)NOUT) { bv = (unsigned int)b; break; } }
    thr[1] = bv << 21;
  }
}

// per-(head, channel) max|dense| -> scale (outliers excluded; V sinks excluded)
__global__ __launch_bounds__(256)
void scale_kernel(const float* __restrict__ k, const float* __restrict__ v,
                  const unsigned int* __restrict__ thr,
                  float* __restrict__ sk, float* __restrict__ sv) {
  __shared__ float red[256];
  int id = blockIdx.x;              // 0..8191 : t*4096 + h*128 + d
  int t  = id >> 12;
  int hd = id & 4095;
  int h  = hd >> 7, d = hd & 127;
  const float* src = t ? v : k;
  unsigned int tb = t ? thr[1] : thr[0];
  float mx = 0.f;
  for (int s = threadIdx.x; s < SEQ; s += 256) {
    float x = src[((h * SEQ) + s) * DIM + d];
    unsigned int bits = __float_as_uint(fabsf(x));
    bool excl = (bits >= tb) || (t && s < SINK);
    if (!excl) mx = fmaxf(mx, fabsf(x));
  }
  red[threadIdx.x] = mx;
  __syncthreads();
  for (int o = 128; o >= 1; o >>= 1) {
    if (threadIdx.x < o) red[threadIdx.x] = fmaxf(red[threadIdx.x], red[threadIdx.x + o]);
    __syncthreads();
  }
  if (threadIdx.x == 0) (t ? sv : sk)[hd] = fmaxf(red[0], 1e-6f) / QMAXF;
}

__device__ __forceinline__ float qdq(float x, float sc) {
  float q = fminf(fmaxf(rintf(x / sc), -8.f), QMAXF);  // RNE like jnp.round
  return q * sc;
}

// K: quant-dequant dense, keep outliers exact; store TRANSPOSED [h][d][s] as f16
__global__ void build_k_kernel(const float* __restrict__ k,
                               const unsigned int* __restrict__ thr,
                               const float* __restrict__ sk,
                               _Float16* __restrict__ kt) {
  int i = blockIdx.x * 256 + threadIdx.x;
  if (i >= NTOT) return;
  int d = i & 127; int s = (i >> 7) & 2047; int h = i >> 18;
  float x = k[i];
  float val = (__float_as_uint(fabsf(x)) >= thr[0]) ? x : qdq(x, sk[h * DIM + d]);
  kt[((h * DIM) + d) * SEQ + s] = (_Float16)val;
}

// V: quant-dequant dense, keep outliers + sink tokens exact; row-major f16
__global__ void build_v_kernel(const float* __restrict__ v,
                               const unsigned int* __restrict__ thr,
                               const float* __restrict__ sv,
                               _Float16* __restrict__ vr) {
  int i = blockIdx.x * 256 + threadIdx.x;
  if (i >= NTOT) return;
  int d = i & 127; int s = (i >> 7) & 2047; int h = i >> 18;
  float x = v[i];
  bool keep = (__float_as_uint(fabsf(x)) >= thr[1]) || (s < SINK);
  vr[i] = (_Float16)(keep ? x : qdq(x, sv[h * DIM + d]));
}

// Q -> f16, pre-scaled by 1/sqrt(D), packed in 16x16x32 A-fragment lane order:
//   Qpack[ (((h*128 + mt)*4 + c)*32 + L)*16 + i ]
//   lane L: row = L%16, g = L/16;  half i=2j+b maps to K = (j/4)*16 + g*8 + (j%4)*2 + b
__global__ void pack_q_kernel(const float* __restrict__ q, _Float16* __restrict__ qp) {
  int i = blockIdx.x * 256 + threadIdx.x;
  if (i >= NTOT) return;
  int ii = i & 15;
  int L  = (i >> 4) & 31;
  int c  = (i >> 9) & 3;
  int mt = (i >> 11) & 127;
  int h  = i >> 18;
  int j = ii >> 1, b = ii & 1;
  int row = L & 15, g = L >> 4;
  int kl = (j >> 2) * 16 + g * 8 + (j & 3) * 2 + b;
  int d = c * 32 + kl;
  int s = mt * 16 + row;
  qp[i] = (_Float16)(q[((h * SEQ) + s) * DIM + d] * SMSCALE);
}

// ---------------- flash attention with WMMA ----------------
// grid = (S/128, H), block = 256 (8 wave32); each wave owns 16 query rows.
__global__ __launch_bounds__(256)
void attn_wmma_kernel(const _Float16* __restrict__ qp,
                      const _Float16* __restrict__ kt,
                      const _Float16* __restrict__ vr,
                      float* __restrict__ out) {
  __shared__ _Float16 pbuf[8][16][32];   // per-wave P transpose staging (C->A layout)

  const int h    = blockIdx.y;
  const int qt   = blockIdx.x;
  const int wave = threadIdx.x >> 5;
  const int lane = threadIdx.x & 31;
  const int g    = lane >> 4;           // half-wave group
  const int nlo  = lane & 15;
  const int mt   = qt * 8 + wave;       // global 16-row tile
  const int qbase = mt * 16;

  // Q A-fragments (D=128 -> 4 chunks of 32); one contiguous 32B load per lane
  v16h aq[4];
#pragma unroll
  for (int c = 0; c < 4; ++c)
    aq[c] = *(const v16h*)&qp[((((h * 128) + mt) * 4 + c) * 32 + lane) * 16];

  const v8f vzero = {0.f, 0.f, 0.f, 0.f, 0.f, 0.f, 0.f, 0.f};
  v8f acc[8];
#pragma unroll
  for (int dc = 0; dc < 8; ++dc) acc[dc] = vzero;
  float mrun[8], lsum[8], corr[8];
#pragma unroll
  for (int r = 0; r < 8; ++r) { mrun[r] = -3.0e38f; lsum[r] = 0.f; }

  const int kend = qbase + 16;          // causal: keys 0..qbase+15
  for (int kb = 0; kb < kend; kb += 32) {
    // ---- S = Q * K^T (two N=16 fragments, K-dim = 128 = 4 x 32) ----
    v8f s0 = vzero, s1 = vzero;
#pragma unroll
    for (int c = 0; c < 4; ++c) {
      const _Float16* krow = &kt[((h * DIM) + c * 32 + lane) * SEQ + kb];
      if (kb + 32 < kend) __builtin_prefetch(krow + 32, 0, 3);  // next K tile
      v16h b0, b1;                                  // col = key kb(+16)+nlo, elem s = d (table order)
#pragma unroll
      for (int s = 0; s < 16; ++s) {
        int kk = (s < 8) ? (g * 8 + s) : (16 + g * 8 + (s - 8));
        const _Float16* kp = &kt[((h * DIM) + c * 32 + kk) * SEQ + kb + nlo];
        b0[s] = kp[0];
        b1[s] = kp[16];
      }
      s0 = __builtin_amdgcn_wmma_f32_16x16x32_f16(false, aq[c], false, b0,
                                                  (short)0, s0, false, false);
      s1 = __builtin_amdgcn_wmma_f32_16x16x32_f16(false, aq[c], false, b1,
                                                  (short)0, s1, false, false);
    }

    // ---- online softmax (C-frag: lane holds rows g*8+r at column nlo) ----
#pragma unroll
    for (int r = 0; r < 8; ++r) {
      int m = qbase + g * 8 + r;
      float t0 = (kb + nlo      <= m) ? s0[r] : -3.0e38f;
      float t1 = (kb + 16 + nlo <= m) ? s1[r] : -3.0e38f;
      float mx = fmaxf(t0, t1);
#pragma unroll
      for (int off = 8; off >= 1; off >>= 1)
        mx = fmaxf(mx, __shfl_xor(mx, off, 16));
      float nm = fmaxf(mrun[r], mx);
      float cr = __expf(mrun[r] - nm);
      float p0 = __expf(t0 - nm);
      float p1 = __expf(t1 - nm);
      float rs = p0 + p1;
#pragma unroll
      for (int off = 8; off >= 1; off >>= 1)
        rs += __shfl_xor(rs, off, 16);
      lsum[r] = lsum[r] * cr + rs;
      mrun[r] = nm;
      corr[r] = cr;
      pbuf[wave][g * 8 + r][nlo]      = (_Float16)p0;   // P[m][k_local]
      pbuf[wave][g * 8 + r][nlo + 16] = (_Float16)p1;
    }
#pragma unroll
    for (int dc = 0; dc < 8; ++dc)
#pragma unroll
      for (int r = 0; r < 8; ++r) acc[dc][r] *= corr[r];

    // ---- repack P as 16x32 A-fragment (same-wave LDS, no barrier needed) ----
    v16h pa;
#pragma unroll
    for (int j = 0; j < 8; ++j) {
      int k0 = (j >> 2) * 16 + g * 8 + (j & 3) * 2;
      pa[2 * j]     = pbuf[wave][nlo][k0];
      pa[2 * j + 1] = pbuf[wave][nlo][k0 + 1];
    }

    // ---- O += P * V (B-frag: col = d dc*16+nlo, elem s = key in table order) ----
#pragma unroll
    for (int dc = 0; dc < 8; ++dc) {
      v16h bv;
#pragma unroll
      for (int s = 0; s < 16; ++s) {
        int kk = (s < 8) ? (g * 8 + s) : (16 + g * 8 + (s - 8));
        bv[s] = vr[((h * SEQ) + kb + kk) * DIM + dc * 16 + nlo];
      }
      acc[dc] = __builtin_amdgcn_wmma_f32_16x16x32_f16(false, pa, false, bv,
                                                       (short)0, acc[dc], false, false);
    }
  }

  // ---- epilogue: normalize and store ----
#pragma unroll
  for (int dc = 0; dc < 8; ++dc)
#pragma unroll
    for (int r = 0; r < 8; ++r) {
      int m = qbase + g * 8 + r;
      out[((h * SEQ) + m) * DIM + dc * 16 + nlo] = acc[dc][r] / lsum[r];
    }
}

// ---------------- launcher ----------------
extern "C" void kernel_launch(void* const* d_in, const int* in_sizes, int n_in,
                              void* d_out, int out_size, void* d_ws, size_t ws_size,
                              hipStream_t stream) {
  const float* q = (const float*)d_in[0];
  const float* k = (const float*)d_in[1];
  const float* v = (const float*)d_in[2];
  float* out = (float*)d_out;
  char* ws = (char*)d_ws;

  unsigned int* histk = (unsigned int*)(ws + 0);
  unsigned int* histv = (unsigned int*)(ws + 8192);
  unsigned int* thr   = (unsigned int*)(ws + 16384);
  float* sk = (float*)(ws + 16448);
  float* sv = (float*)(ws + 32832);
  _Float16* kt = (_Float16*)(ws + 65536);
  _Float16* vr = (_Float16*)(ws + 65536 + (size_t)(16u << 20));
  _Float16* qp = (_Float16*)(ws + 65536 + (size_t)(32u << 20));

  const int nblk = NTOT / 256;

  zero_ws_kernel<<<(4112 + 255) / 256, 256, 0, stream>>>((unsigned int*)ws, 4112);
  hist_kernel<<<nblk, 256, 0, stream>>>(k, histk, NTOT);
  hist_kernel<<<nblk, 256, 0, stream>>>(v, histv, NTOT);
  select_thr_kernel<<<1, 32, 0, stream>>>(histk, histv, thr);
  scale_kernel<<<2 * BH * DIM, 256, 0, stream>>>(k, v, thr, sk, sv);
  build_k_kernel<<<nblk, 256, 0, stream>>>(k, thr, sk, kt);
  build_v_kernel<<<nblk, 256, 0, stream>>>(v, thr, sv, vr);
  pack_q_kernel<<<nblk, 256, 0, stream>>>(q, qp);

  dim3 grid(SEQ / 128, BH);
  attn_wmma_kernel<<<grid, 256, 0, stream>>>(qp, kt, vr, out);
}
